// LSHPromptSelector_79963701116972
// MI455X (gfx1250) — matrix operand order repaired and checked by reading, not compiled, not run on hardware
//
#include <hip/hip_runtime.h>

typedef float v8f __attribute__((ext_vector_type(8)));
typedef __bf16 v16bf __attribute__((ext_vector_type(16)));
typedef unsigned short u16x8 __attribute__((ext_vector_type(8)));

union BF16Frag {
    v16bf v;
    u16x8 h[2];
    unsigned short u[16];
};

#define B_ROWS 16384
#define D_EXP  8192
#define K_DIM  768
#define TOPK   409
#define NSEL   8
#define CHUNK  1024
#define POOLF  30.0f
#define HASHM  2654435761.0f

// LDS layout (bytes)
#define EXP_OFF   0            // 16*8192 bf16 bits  = 262144
#define A_OFF     262144       // 16*768  bf16 bits  = 24576
#define HIST_OFF  286720       // 16*256  u32        = 16384
#define BSTAR_OFF 303104       // 16 u32
#define NEED_OFF  303168       // 16 u32
#define THR_OFF   303232       // 16 u32
#define PART_OFF  303296       // 256 f32            = 1024
#define LDS_BYTES 304320

__device__ __forceinline__ unsigned short f2bf(float f) {
    __bf16 b = (__bf16)f;                      // native v_cvt path
    return __builtin_bit_cast(unsigned short, b);
}
__device__ __forceinline__ float bf2f(unsigned short u) {
    __bf16 b = __builtin_bit_cast(__bf16, u);
    return (float)b;
}
__device__ __forceinline__ unsigned int ord16(unsigned short u) {
    // monotone map: larger ord <=> larger bf16 value
    return (u & 0x8000u) ? (unsigned int)(u ^ 0xFFFFu)
                         : (unsigned int)(u | 0x8000u);
}
__device__ __forceinline__ v8f wmma_bf16(const BF16Frag& a, const BF16Frag& b, v8f c) {
    return __builtin_amdgcn_wmma_f32_16x16x32_bf16(
               false, a.v, false, b.v, (short)0, c, false, false);
}

// ---- shared top-k + hash tail (phases 2/3) ----
__device__ void topk_hash_phase(char* smem, int tid, int row0, int* __restrict__ out) {
    unsigned short* expT  = (unsigned short*)(smem + EXP_OFF);
    unsigned int*   hist  = (unsigned int*)(smem + HIST_OFF);
    unsigned int*   bstar = (unsigned int*)(smem + BSTAR_OFF);
    unsigned int*   need  = (unsigned int*)(smem + NEED_OFF);
    unsigned int*   thr   = (unsigned int*)(smem + THR_OFF);
    float*          part  = (float*)(smem + PART_OFF);

    for (int i = tid; i < 16 * 256; i += 256) hist[i] = 0u;
    __syncthreads();
    {
        const int row = tid >> 4, sub = tid & 15;
        const unsigned short* er = expT + (size_t)row * D_EXP;
        for (int blk = sub; blk < D_EXP / 8; blk += 16) {
            u16x8 v = *reinterpret_cast<const u16x8*>(er + blk * 8);
#pragma unroll
            for (int j = 0; j < 8; ++j) {
                unsigned int o = ord16(v[j]);
                atomicAdd(&hist[row * 256 + (o >> 8)], 1u);
            }
        }
    }
    __syncthreads();
    if (tid < 16) {
        unsigned int cum = 0;
        for (int b = 255; b >= 0; --b) {
            unsigned int h = hist[tid * 256 + b];
            if (cum + h >= TOPK) { bstar[tid] = (unsigned int)b; need[tid] = TOPK - cum; break; }
            cum += h;
        }
    }
    __syncthreads();
    for (int i = tid; i < 16 * 256; i += 256) hist[i] = 0u;
    __syncthreads();
    {
        const int row = tid >> 4, sub = tid & 15;
        const unsigned int bs = bstar[row];
        const unsigned short* er = expT + (size_t)row * D_EXP;
        for (int blk = sub; blk < D_EXP / 8; blk += 16) {
            u16x8 v = *reinterpret_cast<const u16x8*>(er + blk * 8);
#pragma unroll
            for (int j = 0; j < 8; ++j) {
                unsigned int o = ord16(v[j]);
                if ((o >> 8) == bs) atomicAdd(&hist[row * 256 + (o & 255u)], 1u);
            }
        }
    }
    __syncthreads();
    if (tid < 16) {
        unsigned int cum = 0, ls = 0, nd = need[tid];
        for (int l = 255; l >= 0; --l) {
            unsigned int h = hist[tid * 256 + l];
            if (cum + h >= nd) { ls = (unsigned int)l; break; }
            cum += h;
        }
        thr[tid] = (bstar[tid] << 8) | ls;
    }
    __syncthreads();
    {
        const int pair = tid >> 1, ph = tid & 1;
        const int row = pair >> 3, s = pair & 7;
        const unsigned int t = thr[row];
        const unsigned short* base = expT + (size_t)row * D_EXP + s * CHUNK;
        float acc = 0.0f;
        for (int j8 = 0; j8 < 64; ++j8) {
            int cc0 = ph * (CHUNK / 2) + j8 * 8;
            u16x8 v = *reinterpret_cast<const u16x8*>(base + cc0);
#pragma unroll
            for (int j = 0; j < 8; ++j) {
                int cc = cc0 + j;
                unsigned int o = ord16(v[j]);
                float w = fmodf((float)(cc + 1) * HASHM, POOLF);
                acc += (o >= t) ? bf2f(v[j]) * w : 0.0f;
            }
        }
        part[pair * 2 + ph] = acc;
    }
    __syncthreads();
    if (tid < 128) {
        const int row = tid >> 3, s = tid & 7;
        float h = part[tid * 2] + part[tid * 2 + 1];
        float r = fmodf(h, POOLF);
        if (r < 0.0f) r += POOLF;
        out[(size_t)(row0 + row) * NSEL + s] = (int)r;
    }
}

// ---- one-shot f32 -> bf16 conversion into workspace ----
extern "C" __global__ void __launch_bounds__(256)
cvt_f32_bf16(const float* __restrict__ src, unsigned short* __restrict__ dst) {
    int i = (blockIdx.x * 256 + threadIdx.x) * 8;
    const float4* p = reinterpret_cast<const float4*>(src + i);
    float4 a = p[0], b = p[1];
    u16x8 o;
    o[0] = f2bf(a.x); o[1] = f2bf(a.y); o[2] = f2bf(a.z); o[3] = f2bf(a.w);
    o[4] = f2bf(b.x); o[5] = f2bf(b.y); o[6] = f2bf(b.z); o[7] = f2bf(b.w);
    *reinterpret_cast<u16x8*>(dst + i) = o;
}

// ---- main fused kernel, bf16 sources in workspace ----
extern "C" __global__ void __launch_bounds__(256, 1)
lsh_main_bf16(const unsigned short* __restrict__ Qbf,
              const unsigned short* __restrict__ Pbf,
              int* __restrict__ out) {
    extern __shared__ char smem[];
    unsigned short* expT = (unsigned short*)(smem + EXP_OFF);
    unsigned short* aLds = (unsigned short*)(smem + A_OFF);

    const int tid  = threadIdx.x;
    const int lane = tid & 31;
    const int wave = tid >> 5;
    const int hlf  = lane >> 4;
    const int l15  = lane & 15;
    const int row0 = blockIdx.x * 16;

    // Stage A tile (16 x 768 bf16 = 1536 x 16B) global -> LDS via async copy
    for (int c = tid; c < 1536; c += 256) {
        int m = c / 96, off = (c % 96) * 8;                 // 96 chunks/row, 8 u16 per chunk
        unsigned long long g = (unsigned long long)(Qbf + (size_t)(row0 + m) * K_DIM + off);
        unsigned int l = (unsigned int)(size_t)(aLds + m * K_DIM + off);
        asm volatile("global_load_async_to_lds_b128 %0, %1, off"
                     :: "v"(l), "v"(g) : "memory");
    }
    asm volatile("s_wait_asynccnt 0" ::: "memory");
    __syncthreads();

    // GEMM: each wave owns tile-pairs p -> N-tiles {2p, 2p+1}
    for (int p = wave; p < D_EXP / 32; p += 8) {
        const int nbase = p * 32;
        const unsigned short* b0p = Pbf + (size_t)(nbase + l15) * K_DIM;
        const unsigned short* b1p = Pbf + (size_t)(nbase + 16 + l15) * K_DIM;
        v8f c0 = 0.0f, c1 = 0.0f;
#pragma unroll 4
        for (int k0 = 0; k0 < K_DIM; k0 += 32) {
            BF16Frag a, b0, b1;
            a.h[0] = *reinterpret_cast<const u16x8*>(&aLds[l15 * K_DIM + k0 + hlf * 16]);
            a.h[1] = *reinterpret_cast<const u16x8*>(&aLds[l15 * K_DIM + k0 + hlf * 16 + 8]);
            const u16x8* q0 = reinterpret_cast<const u16x8*>(b0p + k0 + hlf * 16);
            const u16x8* q1 = reinterpret_cast<const u16x8*>(b1p + k0 + hlf * 16);
            b0.h[0] = q0[0]; b0.h[1] = q0[1];
            b1.h[0] = q1[0]; b1.h[1] = q1[1];
            c0 = wmma_bf16(a, b0, c0);
            c1 = wmma_bf16(a, b1, c1);
        }
#pragma unroll
        for (int r = 0; r < 8; ++r) {
            int m = r + hlf * 8;
            expT[(size_t)m * D_EXP + nbase + l15]      = f2bf(c0[r]);
            expT[(size_t)m * D_EXP + nbase + 16 + l15] = f2bf(c1[r]);
        }
    }
    __syncthreads();
    topk_hash_phase(smem, tid, row0, out);
}

// ---- fallback: on-the-fly conversion from f32 sources (no workspace needed) ----
extern "C" __global__ void __launch_bounds__(256, 1)
lsh_main_f32(const float* __restrict__ query,
             const float* __restrict__ proj,
             int* __restrict__ out) {
    extern __shared__ char smem[];
    unsigned short* expT = (unsigned short*)(smem + EXP_OFF);
    unsigned short* aLds = (unsigned short*)(smem + A_OFF);

    const int tid  = threadIdx.x;
    const int lane = tid & 31;
    const int wave = tid >> 5;
    const int hlf  = lane >> 4;
    const int l15  = lane & 15;
    const int row0 = blockIdx.x * 16;

    for (int m = 0; m < 16; ++m) {
        const float* qr = query + (size_t)(row0 + m) * K_DIM;
        for (int c = tid; c < K_DIM; c += 256)
            aLds[m * K_DIM + c] = f2bf(qr[c]);
    }
    __syncthreads();

    for (int nt = wave; nt < D_EXP / 16; nt += 8) {
        const int nbase = nt * 16;
        const float* bcol = proj + (size_t)(nbase + l15) * K_DIM;
        v8f c = 0.0f;
        for (int k0 = 0; k0 < K_DIM; k0 += 32) {
            BF16Frag a, b;
            a.h[0] = *reinterpret_cast<const u16x8*>(&aLds[l15 * K_DIM + k0 + hlf * 16]);
            a.h[1] = *reinterpret_cast<const u16x8*>(&aLds[l15 * K_DIM + k0 + hlf * 16 + 8]);
            const float4* bp = reinterpret_cast<const float4*>(bcol + k0 + hlf * 16);
            float4 f0 = bp[0], f1 = bp[1], f2 = bp[2], f3 = bp[3];
            b.v[0]  = (__bf16)f0.x; b.v[1]  = (__bf16)f0.y; b.v[2]  = (__bf16)f0.z; b.v[3]  = (__bf16)f0.w;
            b.v[4]  = (__bf16)f1.x; b.v[5]  = (__bf16)f1.y; b.v[6]  = (__bf16)f1.z; b.v[7]  = (__bf16)f1.w;
            b.v[8]  = (__bf16)f2.x; b.v[9]  = (__bf16)f2.y; b.v[10] = (__bf16)f2.z; b.v[11] = (__bf16)f2.w;
            b.v[12] = (__bf16)f3.x; b.v[13] = (__bf16)f3.y; b.v[14] = (__bf16)f3.z; b.v[15] = (__bf16)f3.w;
            c = wmma_bf16(a, b, c);
        }
#pragma unroll
        for (int r = 0; r < 8; ++r) {
            int m = r + hlf * 8;
            expT[(size_t)m * D_EXP + nbase + l15] = f2bf(c[r]);
        }
    }
    __syncthreads();
    topk_hash_phase(smem, tid, row0, out);
}

extern "C" void kernel_launch(void* const* d_in, const int* in_sizes, int n_in,
                              void* d_out, int out_size, void* d_ws, size_t ws_size,
                              hipStream_t stream) {
    (void)in_sizes; (void)n_in; (void)out_size;
    const float* query = (const float*)d_in[0];
    const float* proj  = (const float*)d_in[1];
    int* out = (int*)d_out;

    const size_t pElems = (size_t)D_EXP * K_DIM;    // 6,291,456
    const size_t qElems = (size_t)B_ROWS * K_DIM;   // 12,582,912
    const size_t needWs = (pElems + qElems) * 2;    // ~36 MB

    if (ws_size >= needWs) {
        unsigned short* Pbf = (unsigned short*)d_ws;
        unsigned short* Qbf = Pbf + pElems;
        cvt_f32_bf16<<<dim3((unsigned)(pElems / 2048)), dim3(256), 0, stream>>>(proj, Pbf);
        cvt_f32_bf16<<<dim3((unsigned)(qElems / 2048)), dim3(256), 0, stream>>>(query, Qbf);
        lsh_main_bf16<<<dim3(B_ROWS / 16), dim3(256), LDS_BYTES, stream>>>(Qbf, Pbf, out);
    } else {
        lsh_main_f32<<<dim3(B_ROWS / 16), dim3(256), LDS_BYTES, stream>>>(query, proj, out);
    }
}
